// MLPEdgePredictor_9869834846315
// MI455X (gfx1250) — matrix operand order repaired and checked by reading, not compiled, not run on hardware
//
#include <hip/hip_runtime.h>
#include <hip/hip_bf16.h>

typedef __attribute__((ext_vector_type(16))) _Float16 v16h;
typedef __attribute__((ext_vector_type(8)))  _Float16 v8h;
typedef __attribute__((ext_vector_type(8)))  float    v8f;
typedef __attribute__((ext_vector_type(4)))  float    v4f;

#define D        128
#define TWO_D    256
#define TILE     16
#define NWAVES   8
#define LN_EPS   1e-5f
#define ROWPAD   (TWO_D + 8)   // 264 halves = 528B row stride -> conflict-free b128 LDS loads

__launch_bounds__(256)
__global__ void edge_mlp_wmma_kernel(const float* __restrict__ x,
                                     const int*   __restrict__ src,
                                     const int*   __restrict__ dst,
                                     const float* __restrict__ a,
                                     const float* __restrict__ W1,
                                     const float* __restrict__ b1,
                                     const float* __restrict__ W2,
                                     const float* __restrict__ b2,
                                     const float* __restrict__ gamma,
                                     const float* __restrict__ beta,
                                     float* __restrict__ out,
                                     int E, int nTiles)
{
    __shared__ _Float16 sH[TILE][ROWPAD];     // layernormed edge rows, f16
    __shared__ float    sPart[NWAVES][TILE];  // per-wave partial dot products

    const int  tid  = threadIdx.x;
    const int  wv   = tid >> 5;     // wave id 0..7 -> owns W1 columns [16*wv, 16*wv+16)
    const int  lane = tid & 31;
    const bool hi   = lane >= 16;
    const int  l16  = lane & 15;

    // ---- per-lane constants for the second layer ----
    const int   ncol = wv * 16 + l16;      // this lane's output column of layer 1
    const float b1v  = b1[ncol];
    const float w2v  = W2[ncol];
    const float b2v  = b2[0];

    // ---- per-lane LayerNorm affine params (concat columns 8*lane .. 8*lane+7) ----
    const v4f gamV0 = *(const v4f*)&gamma[8 * lane];
    const v4f gamV1 = *(const v4f*)&gamma[8 * lane + 4];
    const v4f betV0 = *(const v4f*)&beta [8 * lane];
    const v4f betV1 = *(const v4f*)&beta [8 * lane + 4];

    // ---- preload this wave's W1 slice as WMMA B fragments (f32 -> f16) ----
    // B (32x16 f16) k order used here (same as A): lanes 0-15 hold K=0..15 of column N=lane, lanes 16-31
    // hold K=16..31 of column N=lane-16; VGPR v packs K=2v (lo16) and K=2v+1 (hi16).
    v16h bF[8];
    #pragma unroll
    for (int kb = 0; kb < 8; ++kb) {
        v16h bb;
        #pragma unroll
        for (int v = 0; v < 8; ++v) {
            int k = kb * 32 + (hi ? 16 : 0) + 2 * v;
            bb[2 * v]     = (_Float16)W1[(size_t)k       * D + ncol];
            bb[2 * v + 1] = (_Float16)W1[(size_t)(k + 1) * D + ncol];
        }
        bF[kb] = bb;
    }

    const int tileStride = gridDim.x;

    for (int tile = blockIdx.x; tile < nTiles; tile += tileStride) {
        // ================= phase 1: gather + scale + layernorm -> LDS (f16) =========
        // Each wave produces 2 edge rows; lane covers concat columns [8*lane, 8*lane+8).
        #pragma unroll
        for (int tt = 0; tt < 2; ++tt) {
            const int t = wv * 2 + tt;           // row in tile 0..15
            const int e = tile * TILE + t;
            v4f v0 = {}, v1 = {};
            if (e < E) {
                const int node = (lane < 16) ? src[e] : dst[e];
                const float* row = x + (size_t)node * D + 8 * (size_t)l16;
                v0 = *(const v4f*)row;
                v1 = *(const v4f*)(row + 4);
                const float ae = a[e];
                v0 *= ae; v1 *= ae;
            }
            // prefetch next grid-stride tile's gather rows (hides L2 latency)
            const int eN = e + tileStride * TILE;
            if (eN < E) {
                const int nodeN = (lane < 16) ? src[eN] : dst[eN];
                __builtin_prefetch(x + (size_t)nodeN * D + 8 * (size_t)l16, 0, 1);
            }
            float s1 = v0[0] + v0[1] + v0[2] + v0[3] + v1[0] + v1[1] + v1[2] + v1[3];
            float s2 = v0[0]*v0[0] + v0[1]*v0[1] + v0[2]*v0[2] + v0[3]*v0[3]
                     + v1[0]*v1[0] + v1[1]*v1[1] + v1[2]*v1[2] + v1[3]*v1[3];
            #pragma unroll
            for (int m = 16; m >= 1; m >>= 1) {   // full wave32 butterfly reduce
                s1 += __shfl_xor(s1, m, 32);
                s2 += __shfl_xor(s2, m, 32);
            }
            const float mu   = s1 * (1.f / TWO_D);
            const float var  = s2 * (1.f / TWO_D) - mu * mu;
            const float rstd = rsqrtf(var + LN_EPS);
            v8h hv;
            #pragma unroll
            for (int j = 0; j < 4; ++j) {
                hv[j]     = (_Float16)((v0[j] - mu) * rstd * gamV0[j] + betV0[j]);
                hv[4 + j] = (_Float16)((v1[j] - mu) * rstd * gamV1[j] + betV1[j]);
            }
            *(v8h*)&sH[t][8 * lane] = hv;         // 16B aligned, conflict-free
        }
        __syncthreads();

        // ================= phase 2: z-tile = LN(h) @ W1-slice via WMMA ==============
        // A (16x32 f16) layout: lane holds row M=lane%16; halves K = kb*32 +
        // (hi?16:0)+[0..7] in VGPRs 0-3 and +8 more in VGPRs 4-7.
        // 2-stage software pipeline: WMMA for k-step kb consumes the fragment
        // loaded in step kb-1, so the ds_load_b128 pair for kb+1 can issue
        // before (and overlap with) the WMMA for kb.
        const _Float16* sRow = &sH[l16][0];
        const int aoff = hi ? 16 : 0;
        v8h a0 = *(const v8h*)&sRow[aoff];
        v8h a1 = *(const v8h*)&sRow[aoff + 8];
        v8f c = {};
        #pragma unroll
        for (int kb = 0; kb < 8; ++kb) {
            v16h afc;
            #pragma unroll
            for (int i = 0; i < 8; ++i) { afc[i] = a0[i]; afc[8 + i] = a1[i]; }
            if (kb < 7) {                         // issue next loads before WMMA
                const int off = (kb + 1) * 32 + aoff;
                a0 = *(const v8h*)&sRow[off];
                a1 = *(const v8h*)&sRow[off + 8];
            }
            c = __builtin_amdgcn_wmma_f32_16x16x32_f16(
                    /*neg_a=*/false, afc, /*neg_b=*/false, bF[kb],
                    /*c_mod=*/(short)0, c, /*reuse_a=*/false, /*reuse_b=*/false);
        }

        // ================= phase 3: +b1, branchless ELU, dot with W2 slice ==========
        // C frag: VGPR r holds row M = r + (hi?8:0), column N = ncol.
        // elu(z) = max(z,0) + (exp(min(z,0)) - 1)  -> one v_exp_f32, no EXEC branch
        float p[8];
        #pragma unroll
        for (int r = 0; r < 8; ++r) {
            const float z  = c[r] + b1v;
            const float zp = fmaxf(z, 0.f);
            const float zn = fminf(z, 0.f);
            p[r] = (zp + (__expf(zn) - 1.f)) * w2v;
        }
        #pragma unroll
        for (int m = 1; m < 16; m <<= 1) {        // reduce over N within 16-lane halves
            #pragma unroll
            for (int r = 0; r < 8; ++r) p[r] += __shfl_xor(p[r], m, 32);
        }
        if (l16 == 0) {
            const int rowBase = hi ? 8 : 0;
            #pragma unroll
            for (int r = 0; r < 8; ++r) sPart[wv][rowBase + r] = p[r];
        }
        __syncthreads();

        // ================= phase 4: cross-wave reduce + store =======================
        if (tid < TILE) {
            const int e = tile * TILE + tid;
            if (e < E) {
                float s = 0.f;
                #pragma unroll
                for (int w = 0; w < NWAVES; ++w) s += sPart[w][tid];
                out[e] = s + b2v;
            }
        }
        __syncthreads();   // protect sH / sPart before next tile
    }
}

extern "C" void kernel_launch(void* const* d_in, const int* in_sizes, int n_in,
                              void* d_out, int out_size, void* d_ws, size_t ws_size,
                              hipStream_t stream) {
    const float* x     = (const float*)d_in[0];
    const int*   src   = (const int*)  d_in[1];
    const int*   dst   = (const int*)  d_in[2];
    const float* a     = (const float*)d_in[3];
    const float* W1    = (const float*)d_in[4];
    const float* b1    = (const float*)d_in[5];
    const float* W2    = (const float*)d_in[6];
    const float* b2    = (const float*)d_in[7];
    const float* gamma = (const float*)d_in[8];
    const float* beta  = (const float*)d_in[9];
    float*       out   = (float*)d_out;

    const int E      = in_sizes[1];               // number of edges
    const int nTiles = (E + TILE - 1) / TILE;
    int grid = nTiles < 4096 ? nTiles : 4096;     // grid-stride; amortize W1 preload

    edge_mlp_wmma_kernel<<<grid, 256, 0, stream>>>(
        x, src, dst, a, W1, b1, W2, b2, gamma, beta, out, E, nTiles);
}
